// Fp8Linear_80900003988134
// MI455X (gfx1250) — 8-bit matrix operand order repaired and checked by reading, compiled, not run on hardware
//
#include <hip/hip_runtime.h>
#include <hip/hip_bf16.h>

typedef __attribute__((ext_vector_type(16))) int   v16i;
typedef __attribute__((ext_vector_type(8)))  float v8f;
typedef __attribute__((ext_vector_type(4)))  int   v4i;
typedef __attribute__((ext_vector_type(2)))  int   v2i;

#define GEMM_M 1024
#define GEMM_K 4096
#define GEMM_N 4096
#define F8_MAX 448.0f

#define TILE_ROWS   128
#define TILE_BYTES  128
#define TILE_STRIDE 144
#define TILE_SZ     (TILE_ROWS * TILE_STRIDE)
#define BUF_SZ      (2 * TILE_SZ)

__device__ __forceinline__ unsigned int f32_to_e4m3_sw(float f) {
    unsigned int u    = __float_as_uint(f);
    unsigned int sign = (u >> 24) & 0x80u;
    unsigned int abs  = u & 0x7FFFFFFFu;
    if (abs < 0x3C800000u) {
        float a = __uint_as_float(abs);
        int m = (int)rintf(a * 512.0f);
        if (m >= 8) return sign | 0x08u;
        return sign | (unsigned int)m;
    }
    unsigned int mant = abs & 0x7FFFFFu;
    unsigned int exp  = abs >> 23;
    unsigned int lsb  = (mant >> 20) & 1u;
    mant = (mant + 0x7FFFFu + lsb) >> 20;
    if (mant == 8u) { mant = 0u; exp++; }
    int e8 = (int)exp - 127 + 7;
    if (e8 >= 16) { e8 = 15; mant = 6u; }
    return sign | ((unsigned int)e8 << 3) | mant;
}

__device__ __forceinline__ unsigned int pack4_e4m3(float a, float b, float c, float d) {
#if __has_builtin(__builtin_amdgcn_cvt_pk_fp8_f32)
    int lo = __builtin_amdgcn_cvt_pk_fp8_f32(a, b, 0, false);
    int hi = __builtin_amdgcn_cvt_pk_fp8_f32(c, d, lo, true);
    return (unsigned int)hi;
#else
    return f32_to_e4m3_sw(a) | (f32_to_e4m3_sw(b) << 8) |
           (f32_to_e4m3_sw(c) << 16) | (f32_to_e4m3_sw(d) << 24);
#endif
}

__device__ __forceinline__ void async_copy_b128(unsigned int lds_addr, unsigned long long gaddr) {
    asm volatile("global_load_async_to_lds_b128 %0, %1, off"
                 :: "v"(lds_addr), "v"(gaddr)
                 : "memory");
}

__device__ __forceinline__ void wait_asynccnt0() {
    asm volatile("s_wait_asynccnt 0x0" ::: "memory");
}

__global__ void init_kernel(unsigned int* amax_bits) {
    amax_bits[0] = 0u;
    amax_bits[1] = 0u;
}

__global__ __launch_bounds__(256) void amax_kernel(const float* __restrict__ src, int n,
                                                   unsigned int* __restrict__ amax_bits) {
    int i      = blockIdx.x * blockDim.x + threadIdx.x;
    int stride = gridDim.x * blockDim.x;
    float m = 0.0f;
    for (; i < n; i += stride) m = fmaxf(m, fabsf(src[i]));
    #pragma unroll
    for (int off = 16; off > 0; off >>= 1)
        m = fmaxf(m, __shfl_xor(m, off, 32));
    if ((threadIdx.x & 31) == 0)
        atomicMax(amax_bits, __float_as_uint(m));
}

__global__ void scales_kernel(const unsigned int* __restrict__ amax_bits,
                              float* __restrict__ scales) {
    float ax = fmaxf(__uint_as_float(amax_bits[0]), 1e-12f);
    float aw = fmaxf(__uint_as_float(amax_bits[1]), 1e-12f);
    scales[0] = F8_MAX / ax;
    scales[1] = F8_MAX / aw;
    scales[2] = (ax / F8_MAX) * (aw / F8_MAX);
}

__global__ __launch_bounds__(256) void quantize_kernel(const float* __restrict__ src,
                                                       unsigned int* __restrict__ dst,
                                                       int n4,
                                                       const float* __restrict__ scale_ptr) {
    const float s = *scale_ptr;
    int i      = blockIdx.x * blockDim.x + threadIdx.x;
    int stride = gridDim.x * blockDim.x;
    for (; i < n4; i += stride) {
        float4 x = ((const float4*)src)[i];
        float a = fminf(fmaxf(x.x * s, -F8_MAX), F8_MAX);
        float b = fminf(fmaxf(x.y * s, -F8_MAX), F8_MAX);
        float c = fminf(fmaxf(x.z * s, -F8_MAX), F8_MAX);
        float d = fminf(fmaxf(x.w * s, -F8_MAX), F8_MAX);
        dst[i] = pack4_e4m3(a, b, c, d);
    }
}

__global__ __launch_bounds__(256) void fp8_gemm_kernel(
    const unsigned char* __restrict__ qA,
    const unsigned char* __restrict__ qW,
    const float* __restrict__ scale_ptr,
    const float* __restrict__ bias,
    float* __restrict__ out)
{
    __shared__ __align__(16) unsigned char smem[2 * BUF_SZ];

    const int tid  = threadIdx.x;
    const int wave = tid >> 5;
    const int lane = tid & 31;
    const int half = lane >> 4;
    const int l16  = lane & 15;

    const int nBlocks = GEMM_N / 128;
    const int bm = (blockIdx.x / nBlocks) * 128;
    const int bn = (blockIdx.x % nBlocks) * 128;

    const int wmRel = (wave & 3) * 32;
    const int wnRel = (wave >> 2) * 64;

    const float s = *scale_ptr;

    const int cpRow = tid >> 3;
    const int cpCol = (tid & 7) * 16;
    unsigned int aLds[4], bLds[4];
    unsigned long long aG[4], bG[4];
    #pragma unroll
    for (int r = 0; r < 4; ++r) {
        const int row = cpRow + r * 32;
        aLds[r] = (unsigned int)(unsigned long)(uintptr_t)&smem[row * TILE_STRIDE + cpCol];
        bLds[r] = aLds[r] + TILE_SZ;
        aG[r] = (unsigned long long)(uintptr_t)(qA + (size_t)(bm + row) * GEMM_K + cpCol);
        bG[r] = (unsigned long long)(uintptr_t)(qW + (size_t)(bn + row) * GEMM_K + cpCol);
    }

    v8f acc[2][4];
    #pragma unroll
    for (int i = 0; i < 2; ++i)
        #pragma unroll
        for (int j = 0; j < 4; ++j)
            #pragma unroll
            for (int r = 0; r < 8; ++r)
                acc[i][j][r] = 0.0f;

    #pragma unroll
    for (int r = 0; r < 4; ++r) {
        async_copy_b128(aLds[r], aG[r]);
        async_copy_b128(bLds[r], bG[r]);
    }

    const int KCHUNKS = GEMM_K / 128;
    for (int kc = 0; kc < KCHUNKS; ++kc) {
        wait_asynccnt0();
        __syncthreads();

        if (kc + 1 < KCHUNKS) {
            const unsigned int bufN = ((kc + 1) & 1) * BUF_SZ;
            const unsigned long long gOff = (unsigned long long)(kc + 1) * 128ull;
            #pragma unroll
            for (int r = 0; r < 4; ++r) {
                async_copy_b128(aLds[r] + bufN, aG[r] + gOff);
                async_copy_b128(bLds[r] + bufN, bG[r] + gOff);
            }
        }

        const unsigned char* buf = smem + (kc & 1) * BUF_SZ;

        v16i a[2];
        #pragma unroll
        for (int i = 0; i < 2; ++i) {
            const unsigned char* ap = buf + (wmRel + i * 16 + l16) * TILE_STRIDE + half * 8;
            #pragma unroll
            for (int j = 0; j < 4; ++j) {
                v2i t0 = *(const v2i*)(ap + j * 16);
                a[i][2 * j + 0] = t0.x;
                a[i][2 * j + 1] = t0.y;
                v2i t1 = *(const v2i*)(ap + 64 + j * 16);
                a[i][8 + 2 * j + 0] = t1.x;
                a[i][8 + 2 * j + 1] = t1.y;
            }
        }
        v16i b[4];
        #pragma unroll
        for (int i = 0; i < 4; ++i) {
            const unsigned char* bp = buf + TILE_SZ +
                                      (wnRel + i * 16 + l16) * TILE_STRIDE + half * 8;
            #pragma unroll
            for (int j = 0; j < 4; ++j) {
                v2i t0 = *(const v2i*)(bp + j * 16);
                b[i][2 * j + 0] = t0.x;
                b[i][2 * j + 1] = t0.y;
                v2i t1 = *(const v2i*)(bp + 64 + j * 16);
                b[i][8 + 2 * j + 0] = t1.x;
                b[i][8 + 2 * j + 1] = t1.y;
            }
        }
        #pragma unroll
        for (int i = 0; i < 2; ++i)
            #pragma unroll
            for (int j = 0; j < 4; ++j)
                acc[i][j] = __builtin_amdgcn_wmma_f32_16x16x128_fp8_fp8(
                    a[i], b[j], (short)0, acc[i][j], false, false);
    }

    #pragma unroll
    for (int i = 0; i < 2; ++i) {
        #pragma unroll
        for (int j = 0; j < 4; ++j) {
            const int col = bn + wnRel + j * 16 + l16;
            const float bv = bias[col];
            #pragma unroll
            for (int r = 0; r < 8; ++r) {
                const int row = bm + wmRel + i * 16 + half * 8 + r;
                out[(size_t)row * GEMM_N + col] = acc[i][j][r] * s + bv;
            }
        }
    }
}

extern "C" void kernel_launch(void* const* d_in, const int* in_sizes, int n_in,
                              void* d_out, int out_size, void* d_ws, size_t ws_size,
                              hipStream_t stream) {
    const float* inp  = (const float*)d_in[0];
    const float* wt   = (const float*)d_in[1];
    const float* bias = (const float*)d_in[2];
    float* out        = (float*)d_out;

    unsigned char* ws        = (unsigned char*)d_ws;
    unsigned int*  amax_bits = (unsigned int*)ws;
    float*         scales    = (float*)(ws + 16);
    unsigned char* qA        = ws + 256;
    unsigned char* qW        = qA + (size_t)GEMM_M * GEMM_K;

    init_kernel<<<1, 1, 0, stream>>>(amax_bits);
    amax_kernel<<<512, 256, 0, stream>>>(inp, GEMM_M * GEMM_K, amax_bits + 0);
    amax_kernel<<<1024, 256, 0, stream>>>(wt, GEMM_N * GEMM_K, amax_bits + 1);
    scales_kernel<<<1, 1, 0, stream>>>(amax_bits, scales);
    quantize_kernel<<<1024, 256, 0, stream>>>(inp, (unsigned int*)qA,
                                              GEMM_M * GEMM_K / 4, scales + 0);
    quantize_kernel<<<2048, 256, 0, stream>>>(wt, (unsigned int*)qW,
                                              GEMM_N * GEMM_K / 4, scales + 1);

    const int grid = (GEMM_M / 128) * (GEMM_N / 128);
    fp8_gemm_kernel<<<grid, 256, 0, stream>>>(qA, qW, scales + 2, bias, out);
}
